// SelfAttention_73418170958230
// MI455X (gfx1250) — matrix operand order repaired and checked by reading, not compiled, not run on hardware
//
#include <hip/hip_runtime.h>
#include <math.h>

// ---- problem dims (fixed by reference) ----
constexpr int Bd  = 2;
constexpr int Sd  = 2048;
constexpr int HDd = 2048;
constexpr int NHd = 16;
constexpr int DHd = 128;
constexpr int AHd = NHd * DHd;   // 2048

typedef __attribute__((ext_vector_type(16))) _Float16 v16h;
typedef __attribute__((ext_vector_type(8)))  float    v8f;

__device__ inline unsigned lds_addr_of(const void* p) {
    // generic shared addr = {aperture_hi, lds_offset}; truncate to LDS offset
    return (unsigned)(unsigned long long)(uintptr_t)p;
}

// ---------------------------------------------------------------------------
// Tile staging: prefer the Tensor Data Mover (one DMA per tile, TENSORcnt),
// fall back to per-lane global_load_async_to_lds_b128 (ASYNCcnt).
// ---------------------------------------------------------------------------
#if __has_builtin(__builtin_amdgcn_tensor_load_to_lds)
#define USE_TDM 1
#else
#define USE_TDM 0
#endif

#if USE_TDM
typedef unsigned __attribute__((ext_vector_type(4))) uint32x4;
typedef int      __attribute__((ext_vector_type(8))) int32x8;
typedef int      __attribute__((ext_vector_type(4))) int32x4;

// 2D f16 tile load: tile_d1 rows x tile_d0 elems, row stride stride0 elems.
// D# layout per CDNA5 ISA (data_size=1 -> 2B).
__device__ inline void tdm_load_2d(const void* gaddr, unsigned lds_addr,
                                   unsigned tile_d0, unsigned tile_d1,
                                   unsigned long long stride0,
                                   unsigned long long tensor_d0,
                                   unsigned long long tensor_d1) {
    const unsigned long long ga = (unsigned long long)(uintptr_t)gaddr;
    uint32x4 g0;
    g0[0] = 1u;                                    // count=1, user descriptor
    g0[1] = lds_addr;                              // lds_addr [63:32]
    g0[2] = (unsigned)ga;                          // global_addr[31:0]
    g0[3] = (unsigned)((ga >> 32) & 0x01FFFFFFu)   // global_addr[56:32]
          | (2u << 30);                            // type=2 ("image")
    int32x8 g1;
    g1[0] = (int)(1u << 16);                                   // data_size=1 (2B)
    g1[1] = (int)((tensor_d0 & 0xFFFFu) << 16);                // tensor_dim0[15:0]
    g1[2] = (int)(((tensor_d0 >> 16) & 0xFFFFu)
          |       ((tensor_d1 & 0xFFFFu) << 16));              // dim0[31:16]|dim1[15:0]
    g1[3] = (int)(((tensor_d1 >> 16) & 0xFFFFu)
          |       ((tile_d0 & 0xFFFFu) << 16));                // dim1[31:16]|tile0
    g1[4] = (int)(tile_d1 & 0xFFFFu);                          // tile1 (tile2=0)
    g1[5] = (int)(stride0 & 0xFFFFFFFFu);                      // dim0_stride[31:0]
    g1[6] = (int)((stride0 >> 32) & 0xFFFFu);                  // dim0_stride[47:32]
    g1[7] = 0;
    int32x4 z4 = {0, 0, 0, 0};
#if defined(__clang_major__) && (__clang_major__ >= 23)
    int32x8 z8 = {0, 0, 0, 0, 0, 0, 0, 0};
    __builtin_amdgcn_tensor_load_to_lds(g0, g1, z4, z4, z8, 0);
#else
    __builtin_amdgcn_tensor_load_to_lds(g0, g1, z4, z4, 0);
#endif
}
#else  // !USE_TDM : async global->LDS copies
__device__ inline void async_ld_b128(unsigned lds_addr, const void* gaddr) {
    asm volatile("global_load_async_to_lds_b128 %0, %1, off"
                 :: "v"(lds_addr), "v"(gaddr)
                 : "memory");
}
#define WAIT_ASYNC_16() asm volatile("s_wait_asynccnt 16" ::: "memory")
#define WAIT_ASYNC_8()  asm volatile("s_wait_asynccnt 8"  ::: "memory")
#define WAIT_ASYNC_0()  asm volatile("s_wait_asynccnt 0"  ::: "memory")
#endif

// ---------------------------------------------------------------------------
// VALU row reductions via DPP row_ror (no LDS, no DScnt stalls).
// Rotate-butterfly 8,4,2,1 over the 16-lane row holding the N columns.
// ---------------------------------------------------------------------------
template <int R>
__device__ inline float rotf(float x) {
    const int s = __float_as_int(x);
    const int v = __builtin_amdgcn_update_dpp(s, s, 0x120 | R, 0xF, 0xF, false);
    return __int_as_float(v);
}
__device__ inline float row_allmax(float x) {
    x = fmaxf(x, rotf<8>(x));
    x = fmaxf(x, rotf<4>(x));
    x = fmaxf(x, rotf<2>(x));
    x = fmaxf(x, rotf<1>(x));
    return x;
}
__device__ inline float row_allsum(float x) {
    x += rotf<8>(x);
    x += rotf<4>(x);
    x += rotf<2>(x);
    x += rotf<1>(x);
    return x;
}

// ---------------------------------------------------------------------------
// Fragment loaders using contiguous k order k = 16*(lane/16) + s for both A and B
// ---------------------------------------------------------------------------
__device__ inline v16h load_frag_a(const _Float16* __restrict__ p, int ld, int lane) {
    const int g   = lane >> 4;
    const int row = lane & 15;
    const _Float16* r = p + (size_t)row * ld + 16 * g;
    v16h f;
#pragma unroll
    for (int j = 0; j < 8; ++j)  f[j] = r[j];
#pragma unroll
    for (int j = 8; j < 16; ++j) f[j] = r[j];
    return f;
}

// B[k][n] = src[n*ld + k]  (row-major [N,K] source: W^T / K^T).
__device__ inline v16h load_frag_bT(const _Float16* __restrict__ p, int ld, int lane) {
    const int g = lane >> 4;
    const int n = lane & 15;
    const _Float16* r = p + (size_t)n * ld + 16 * g;
    v16h f;
#pragma unroll
    for (int j = 0; j < 16; ++j) f[j] = r[j];
    return f;
}

// B[k][n] = src[k*ld + n]  (row-major [K,N] source: V).
__device__ inline v16h load_frag_bK(const _Float16* __restrict__ p, int ld, int lane) {
    const int g = lane >> 4;
    const int n = lane & 15;
    v16h f;
#pragma unroll
    for (int j = 0; j < 16; ++j) f[j] = p[(size_t)(16 * g + j) * ld + n];
    return f;
}

__device__ inline v8f wmma_f16(v16h a, v16h b, v8f c) {
    return __builtin_amdgcn_wmma_f32_16x16x32_f16(
        false, a, false, b, (short)0, c, false, false);
}

// ---------------------------------------------------------------------------
// fp32 -> fp16 cast (grid-stride)
// ---------------------------------------------------------------------------
__global__ void cast_f32_f16(const float* __restrict__ src, _Float16* __restrict__ dst, size_t n) {
    size_t i = (size_t)blockIdx.x * blockDim.x + threadIdx.x;
    size_t stride = (size_t)gridDim.x * blockDim.x;
    for (; i < n; i += stride) dst[i] = (_Float16)src[i];
}

// ---------------------------------------------------------------------------
// GEMM:  C[M,N] = A[M,K] @ W[N,K]^T + bias[N]
// 128x128 block tile, 4 waves (each 64x64), K-step 64, double-buffered LDS.
// ---------------------------------------------------------------------------
template <bool F32OUT>
__global__ __launch_bounds__(128)
void gemm_wmma(const _Float16* __restrict__ A, const _Float16* __restrict__ W,
               const float* __restrict__ bias, void* __restrict__ Cout,
               int Mdim, int Ndim, int Kdim) {
    __shared__ _Float16 tA[2][128 * 64];   // [row in tile][k]
    __shared__ _Float16 tB[2][128 * 64];   // [col in tile][k]

    const int t    = threadIdx.x;
    const int lane = t & 31;
    const int wave = t >> 5;
    const int wm = wave >> 1, wn = wave & 1;
    const int bm = blockIdx.x * 128, bn = blockIdx.y * 128;
    const int g = lane >> 4, n = lane & 15;

    v8f acc[4][4];
#pragma unroll
    for (int i = 0; i < 4; ++i)
#pragma unroll
        for (int j = 0; j < 4; ++j)
#pragma unroll
            for (int e = 0; e < 8; ++e) acc[i][j][e] = 0.0f;

    auto issue_stage = [&](int buf, int k0) {
#if USE_TDM
        if (wave == 0) {   // one DMA per tile; TENSORcnt tracked by wave 0
            tdm_load_2d(A + (size_t)bm * Kdim + k0, lds_addr_of(&tA[buf][0]),
                        64, 128, (unsigned long long)Kdim,
                        (unsigned long long)Kdim, (unsigned long long)Mdim);
            tdm_load_2d(W + (size_t)bn * Kdim + k0, lds_addr_of(&tB[buf][0]),
                        64, 128, (unsigned long long)Kdim,
                        (unsigned long long)Kdim, (unsigned long long)Ndim);
        }
#else
        const _Float16* ga = A + (size_t)(bm + t) * Kdim + k0;
        const _Float16* gb = W + (size_t)(bn + t) * Kdim + k0;
        const unsigned  la = lds_addr_of(&tA[buf][t * 64]);
        const unsigned  lb = lds_addr_of(&tB[buf][t * 64]);
#pragma unroll
        for (int c = 0; c < 8; ++c) {
            async_ld_b128(la + c * 16, ga + c * 8);
            async_ld_b128(lb + c * 16, gb + c * 8);
        }
#endif
    };

    issue_stage(0, 0);
    int buf = 0;
    for (int k0 = 0; k0 < Kdim; k0 += 64) {
        if (k0 + 64 < Kdim) {
            issue_stage(buf ^ 1, k0 + 64);   // overlap next stage's DMA
#if USE_TDM
            if (wave == 0) __builtin_amdgcn_s_wait_tensorcnt(2);
#else
            WAIT_ASYNC_16();
#endif
        } else {
#if USE_TDM
            if (wave == 0) __builtin_amdgcn_s_wait_tensorcnt(0);
#else
            WAIT_ASYNC_0();
#endif
        }
        __syncthreads();

#pragma unroll
        for (int kc = 0; kc < 64; kc += 32) {
            v16h af[4], bf[4];
#pragma unroll
            for (int i = 0; i < 4; ++i)
                af[i] = load_frag_a(&tA[buf][(wm * 64 + i * 16) * 64 + kc], 64, lane);
#pragma unroll
            for (int j = 0; j < 4; ++j)
                bf[j] = load_frag_bT(&tB[buf][(wn * 64 + j * 16) * 64 + kc], 64, lane);
#pragma unroll
            for (int i = 0; i < 4; ++i)
#pragma unroll
                for (int j = 0; j < 4; ++j)
                    acc[i][j] = wmma_f16(af[i], bf[j], acc[i][j]);
        }
        __syncthreads();
        buf ^= 1;
    }

#pragma unroll
    for (int i = 0; i < 4; ++i)
#pragma unroll
        for (int j = 0; j < 4; ++j) {
            const int col = bn + wn * 64 + j * 16 + n;
            const float bv = bias[col];
#pragma unroll
            for (int r = 0; r < 8; ++r) {
                const int row = bm + wm * 64 + i * 16 + 8 * g + r;
                const float val = acc[i][j][r] + bv;
                if (F32OUT)
                    ((float*)Cout)[(size_t)row * Ndim + col] = val;
                else
                    ((_Float16*)Cout)[(size_t)row * Ndim + col] = (_Float16)val;
            }
        }
}

// ---------------------------------------------------------------------------
// Fused RMSNorm (over AH=2048) + interleaved RoPE, in-place on f16 rows.
// ---------------------------------------------------------------------------
__global__ __launch_bounds__(256)
void rmsnorm_rope_kernel(_Float16* __restrict__ X, const float* __restrict__ gma,
                         const float* __restrict__ fc, const float* __restrict__ fs) {
    const int token = blockIdx.x;
    const int s = token % Sd;
    _Float16* row = X + (size_t)token * AHd;
    const int t = threadIdx.x;

    float v[8];
    float ss = 0.0f;
#pragma unroll
    for (int e = 0; e < 8; ++e) {
        v[e] = (float)row[8 * t + e];
        ss += v[e] * v[e];
    }
#pragma unroll
    for (int off = 16; off > 0; off >>= 1) ss += __shfl_down(ss, off);

    __shared__ float red[8];
    const int w = t >> 5, ln = t & 31;
    if (ln == 0) red[w] = ss;
    __syncthreads();
    if (t == 0) {
        float tot = 0.0f;
#pragma unroll
        for (int i = 0; i < 8; ++i) tot += red[i];
        red[0] = rsqrtf(tot / (float)AHd + 1e-6f);
    }
    __syncthreads();
    const float scale = red[0];

#pragma unroll
    for (int p = 0; p < 4; ++p) {
        const int c0 = 8 * t + 2 * p;
        const int c1 = c0 + 1;
        const int pi = c0 >> 1;
        const int ph = pi & (DHd / 2 - 1);
        const float fcv = fc[(size_t)s * (DHd / 2) + ph];
        const float fsv = fs[(size_t)s * (DHd / 2) + ph];
        const float nr = v[2 * p]     * scale * gma[c0];
        const float ni = v[2 * p + 1] * scale * gma[c1];
        row[c0] = (_Float16)(nr * fcv - ni * fsv);
        row[c1] = (_Float16)(nr * fsv + ni * fcv);
    }
}

// ---------------------------------------------------------------------------
// Flash attention.  Block = 4 waves; each wave owns a 16-row Q tile of one
// (batch, head).  K/V streamed in 32-key tiles through double-buffered LDS,
// online softmax with DPP row reductions.
// ---------------------------------------------------------------------------
__global__ __launch_bounds__(128)
void attention_kernel(const _Float16* __restrict__ Q, const _Float16* __restrict__ K,
                      const _Float16* __restrict__ V, _Float16* __restrict__ O) {
    __shared__ _Float16 tK[2][32 * DHd];    // [key][d]
    __shared__ _Float16 tV[2][32 * DHd];    // [key][d]
    __shared__ _Float16 pbuf[4][16 * 32];   // per-wave P re-layout buffer

    const int t    = threadIdx.x;
    const int lane = t & 31;
    const int wave = t >> 5;
    const int g = lane >> 4, n = lane & 15;
    const int b = blockIdx.z, h = blockIdx.y;
    const int q0 = blockIdx.x * 64 + wave * 16;

    const size_t rowB = (size_t)b * Sd;
    const _Float16* Qp = Q + (rowB + q0) * AHd + h * DHd;
    const _Float16* Kp = K + rowB * AHd + h * DHd;
    const _Float16* Vp = V + rowB * AHd + h * DHd;

    v16h qf[4];
#pragma unroll
    for (int d = 0; d < 4; ++d) qf[d] = load_frag_a(Qp + d * 32, AHd, lane);

    v8f oacc[8];
#pragma unroll
    for (int dd = 0; dd < 8; ++dd)
#pragma unroll
        for (int e = 0; e < 8; ++e) oacc[dd][e] = 0.0f;

    float mrow[8], lrow[8];
#pragma unroll
    for (int r = 0; r < 8; ++r) { mrow[r] = -1e30f; lrow[r] = 0.0f; }

    const float scale = 0.08838834764831845f;   // 1/sqrt(128)

    auto issue_kv = [&](int buf, int kt) {
#if USE_TDM
        if (wave == 0) {
            tdm_load_2d(Kp + (size_t)kt * AHd, lds_addr_of(&tK[buf][0]),
                        DHd, 32, (unsigned long long)AHd,
                        (unsigned long long)DHd, (unsigned long long)Sd);
            tdm_load_2d(Vp + (size_t)kt * AHd, lds_addr_of(&tV[buf][0]),
                        DHd, 32, (unsigned long long)AHd,
                        (unsigned long long)DHd, (unsigned long long)Sd);
        }
#else
        const int r  = t >> 2;
        const int qd = t & 3;
        const _Float16* gk = Kp + (size_t)(kt + r) * AHd + qd * 32;
        const _Float16* gv = Vp + (size_t)(kt + r) * AHd + qd * 32;
        const unsigned  lk = lds_addr_of(&tK[buf][r * DHd + qd * 32]);
        const unsigned  lv = lds_addr_of(&tV[buf][r * DHd + qd * 32]);
#pragma unroll
        for (int c = 0; c < 4; ++c) {
            async_ld_b128(lk + c * 16, gk + c * 8);
            async_ld_b128(lv + c * 16, gv + c * 8);
        }
#endif
    };

    issue_kv(0, 0);
    int buf = 0;
    for (int kt = 0; kt < Sd; kt += 32) {
        if (kt + 32 < Sd) {
            issue_kv(buf ^ 1, kt + 32);
#if USE_TDM
            if (wave == 0) __builtin_amdgcn_s_wait_tensorcnt(2);
#else
            WAIT_ASYNC_8();
#endif
        } else {
#if USE_TDM
            if (wave == 0) __builtin_amdgcn_s_wait_tensorcnt(0);
#else
            WAIT_ASYNC_0();
#endif
        }
        __syncthreads();

        // ---- S = Q @ K^T : 16x32 score tile from LDS K ----
        v8f sf[2];
#pragma unroll
        for (int nk = 0; nk < 2; ++nk) {
#pragma unroll
            for (int e = 0; e < 8; ++e) sf[nk][e] = 0.0f;
#pragma unroll
            for (int d = 0; d < 4; ++d) {
                v16h bf = load_frag_bT(&tK[buf][(nk * 16) * DHd + d * 32], DHd, lane);
                sf[nk] = wmma_f16(qf[d], bf, sf[nk]);
            }
        }

        // ---- online softmax: DPP rotate-reduce over the 16-lane row ----
        float corr[8];
#pragma unroll
        for (int r = 0; r < 8; ++r) {
            const float s0 = sf[0][r] * scale;
            const float s1 = sf[1][r] * scale;
            const float mx = row_allmax(fmaxf(s0, s1));
            const float nm = fmaxf(mrow[r], mx);
            const float p0 = __expf(s0 - nm);
            const float p1 = __expf(s1 - nm);
            corr[r] = __expf(mrow[r] - nm);
            const float rs = row_allsum(p0 + p1);
            lrow[r] = lrow[r] * corr[r] + rs;
            mrow[r] = nm;
            sf[0][r] = p0;
            sf[1][r] = p1;
        }
#pragma unroll
        for (int dd = 0; dd < 8; ++dd)
#pragma unroll
            for (int r = 0; r < 8; ++r) oacc[dd][r] *= corr[r];

        // ---- re-layout P (C layout -> A layout) via per-wave LDS ----
        _Float16* pb = pbuf[wave];
#pragma unroll
        for (int nk = 0; nk < 2; ++nk)
#pragma unroll
            for (int r = 0; r < 8; ++r)
                pb[(8 * g + r) * 32 + nk * 16 + n] = (_Float16)sf[nk][r];
        asm volatile("" ::: "memory");   // same-wave LDS ops are in-order
        v16h pf = load_frag_a(pb, 32, lane);
        asm volatile("" ::: "memory");

        // ---- O += P @ V from LDS V ----
#pragma unroll
        for (int dd = 0; dd < 8; ++dd) {
            v16h bf = load_frag_bK(&tV[buf][dd * 16], DHd, lane);
            oacc[dd] = wmma_f16(pf, bf, oacc[dd]);
        }
        __syncthreads();
        buf ^= 1;
    }

    // ---- finalize: O /= l, store f16 ----
    _Float16* Op = O + (rowB + q0) * AHd + h * DHd;
#pragma unroll
    for (int dd = 0; dd < 8; ++dd)
#pragma unroll
        for (int r = 0; r < 8; ++r) {
            const float val = oacc[dd][r] / lrow[r];
            Op[(size_t)(8 * g + r) * AHd + dd * 16 + n] = (_Float16)val;
        }
}

// ---------------------------------------------------------------------------
// Host-side launcher
// ---------------------------------------------------------------------------
extern "C" void kernel_launch(void* const* d_in, const int* in_sizes, int n_in,
                              void* d_out, int out_size, void* d_ws, size_t ws_size,
                              hipStream_t stream) {
    (void)in_sizes; (void)n_in; (void)out_size; (void)ws_size;

    const float* x  = (const float*)d_in[0];
    const float* fc = (const float*)d_in[1];
    const float* fs = (const float*)d_in[2];
    const float* Wq = (const float*)d_in[3];
    const float* bq = (const float*)d_in[4];
    const float* Wk = (const float*)d_in[5];
    const float* bk = (const float*)d_in[6];
    const float* Wv = (const float*)d_in[7];
    const float* bv = (const float*)d_in[8];
    const float* Wo = (const float*)d_in[9];
    const float* bo = (const float*)d_in[10];
    const float* gq = (const float*)d_in[11];
    const float* gk = (const float*)d_in[12];
    float* out = (float*)d_out;

    const size_t M  = (size_t)Bd * Sd;
    const size_t WN = (size_t)AHd * HDd;

    _Float16* p   = (_Float16*)d_ws;
    _Float16* xh  = p; p += M * HDd;
    _Float16* Wqh = p; p += WN;
    _Float16* Wkh = p; p += WN;
    _Float16* Wvh = p; p += WN;
    _Float16* Woh = p; p += WN;
    _Float16* Qh  = p; p += M * AHd;
    _Float16* Kh  = p; p += M * AHd;
    _Float16* Vh  = p; p += M * AHd;
    _Float16* Oh  = p; p += M * AHd;

    cast_f32_f16<<<2048, 256, 0, stream>>>(x,  xh,  M * HDd);
    cast_f32_f16<<<1024, 256, 0, stream>>>(Wq, Wqh, WN);
    cast_f32_f16<<<1024, 256, 0, stream>>>(Wk, Wkh, WN);
    cast_f32_f16<<<1024, 256, 0, stream>>>(Wv, Wvh, WN);
    cast_f32_f16<<<1024, 256, 0, stream>>>(Wo, Woh, WN);

    dim3 gqkv((unsigned)(M / 128), (unsigned)(AHd / 128));
    gemm_wmma<false><<<gqkv, 128, 0, stream>>>(xh, Wqh, bq, Qh, (int)M, AHd, HDd);
    gemm_wmma<false><<<gqkv, 128, 0, stream>>>(xh, Wkh, bk, Kh, (int)M, AHd, HDd);
    gemm_wmma<false><<<gqkv, 128, 0, stream>>>(xh, Wvh, bv, Vh, (int)M, AHd, HDd);

    rmsnorm_rope_kernel<<<(unsigned)M, 256, 0, stream>>>(Qh, gq, fc, fs);
    rmsnorm_rope_kernel<<<(unsigned)M, 256, 0, stream>>>(Kh, gk, fc, fs);

    attention_kernel<<<dim3(Sd / 64, NHd, Bd), 128, 0, stream>>>(Qh, Kh, Vh, Oh);

    dim3 gout((unsigned)(M / 128), (unsigned)(HDd / 128));
    gemm_wmma<true><<<gout, 128, 0, stream>>>(Oh, Woh, bo, out, (int)M, HDd, AHd);
}
